// MultiHeadAttention_57226144252712
// MI455X (gfx1250) — matrix operand order repaired and checked by reading, not compiled, not run on hardware
//
#include <hip/hip_runtime.h>

// ---------------------------------------------------------------------------
// MI455X (gfx1250) attention forward, bf16 WMMA path, fp32 accumulate.
// B=4, S=2048, D=HD=1024. Outputs: output [B,S,D] then attn [B,S,S] (fp32).
// All GEMMs are A[M,K] x B^T[N,K] (weights pre-transposed, V emitted
// transposed) so both LDS tiles stage with contiguous b128 copies.
// If the toolchain exposes async global->LDS copies (gfx1250 ASYNCcnt path),
// the GEMM uses LDS double-buffering with GLOBAL_LOAD_ASYNC_TO_LDS_B128;
// otherwise it falls back to register-pipelined staging.
// ---------------------------------------------------------------------------

typedef __attribute__((ext_vector_type(16))) __bf16 v16bf;
typedef __attribute__((ext_vector_type(8)))  float  v8f;
typedef int v4i __attribute__((vector_size(16)));   // async-LDS builtin pointee

#define TM 64          // block tile M
#define TN 128         // block tile N
#define TK 32          // K step (one bf16 WMMA depth)
#define LDP 40         // padded LDS row stride (bf16): 80B => 16B-aligned,
                       // conflict-free 4-bank partition for b128 reads

#if defined(__has_builtin)
# if __has_builtin(__builtin_amdgcn_global_load_async_to_lds_b128)
#  define USE_ASYNC_LDS 1
# endif
#endif
#ifndef USE_ASYNC_LDS
# define USE_ASYNC_LDS 0
#endif

#if defined(__has_builtin)
# if __has_builtin(__builtin_amdgcn_s_wait_asynccnt)
#  define WAIT_ASYNC0() __builtin_amdgcn_s_wait_asynccnt(0)
# endif
#endif
#ifndef WAIT_ASYNC0
# define WAIT_ASYNC0() asm volatile("s_wait_asynccnt 0x0" ::: "memory")
#endif

union FragU { uint4 q[2]; v16bf v; };

#if USE_ASYNC_LDS
// one lane: async-copy 16B global -> 16B LDS (ASYNCcnt-tracked, no VGPR data)
// builtin signature (from hipcc diagnostic): (v4i addrspace(1)*, v4i addrspace(3)*, Ii, Ii)
__device__ __forceinline__ void async_b128(const __bf16* g, __bf16* l) {
  __builtin_amdgcn_global_load_async_to_lds_b128(
      (__attribute__((address_space(1))) v4i*)(unsigned long long)g,
      (__attribute__((address_space(3))) v4i*)(unsigned)(unsigned long long)l,
      0, 0);
}
#endif

// shared WMMA micro-step: 1 A fragment reused over 4 B fragments
__device__ __forceinline__ void mma_step(const __bf16* la, const __bf16* lb,
                                         int wr, int wc, int l16, int hlf,
                                         v8f acc[4]) {
  FragU fa;  // A layout: one contiguous 16-elem K run at half*16 (matches B)
  const __bf16* at = la + (wr * 16 + l16) * LDP;
  fa.q[0] = *(const uint4*)(at + hlf * 16);
  fa.q[1] = *(const uint4*)(at + hlf * 16 + 8);
  #pragma unroll
  for (int ns = 0; ns < 4; ++ns) {
    const __bf16* bt = lb + (wc * 64 + ns * 16 + l16) * LDP + hlf * 16;
    FragU fb;  // B layout: one contiguous 16-elem K run at half*16
    fb.q[0] = *(const uint4*)(bt);
    fb.q[1] = *(const uint4*)(bt + 8);
    acc[ns] = __builtin_amdgcn_wmma_f32_16x16x32_bf16(false, fa.v, false, fb.v,
                                                      (short)0, acc[ns], false, false);
  }
}

// ---------------- fp32 -> bf16 conversion (row-major copy) ----------------
__global__ void cvt_f32_bf16(const float* __restrict__ src, __bf16* __restrict__ dst, long n) {
  long i = (long)blockIdx.x * blockDim.x + threadIdx.x;
  if (i < n) dst[i] = (__bf16)src[i];
}

// ------------- fp32 [K,N] -> bf16 [N,K] transposed conversion -------------
__global__ __launch_bounds__(256)
void cvt_transpose_bf16(const float* __restrict__ src, __bf16* __restrict__ dst,
                        int K, int N) {
  __shared__ float tile[32][33];
  const int kb = blockIdx.x * 32, nb = blockIdx.y * 32;
  const int tx = threadIdx.x & 31, ty = threadIdx.x >> 5;
  #pragma unroll
  for (int i = ty; i < 32; i += 8)
    tile[i][tx] = src[(long)(kb + i) * N + nb + tx];
  __syncthreads();
  #pragma unroll
  for (int i = ty; i < 32; i += 8)
    dst[(long)(nb + i) * K + kb + tx] = (__bf16)tile[tx][i];
}

// ---------------------------------------------------------------------------
// Unified WMMA GEMM:  C[M,N] = A[M,K] * Bt[N,K]^T  (+bias) (+residual)
// 256 threads = 8 waves (4x2), each wave computes 16x64 (4 accumulators,
// 4 back-to-back v_wmma_f32_16x16x32_bf16 per K-step, A-fragment reuse).
// SCORES mode: C = (col>row ? 1e-9 : acc*scale), fp32 only.
// OUT_BFT: store bf16 output transposed (used to emit V^T).
// ---------------------------------------------------------------------------
template<bool HAS_BIAS, bool HAS_RES, bool OUT_BF, bool OUT_BFT, bool OUT_F32, bool SCORES>
__global__ __launch_bounds__(256)
void gemm_wmma(const __bf16* __restrict__ A,  const __bf16* __restrict__ Bt,
               const float* __restrict__ bias, const float* __restrict__ resid,
               __bf16* __restrict__ Cbf, float* __restrict__ Cf,
               int M, int N, int K, int lda, int ldb, int ldct,
               long sA, long sB, long sRes, long sC, float scale)
{
  const int bz = blockIdx.z;
  A  += (long)bz * sA;
  Bt += (long)bz * sB;
  const float* res = HAS_RES ? (resid + (long)bz * sRes) : nullptr;
  const long cOff = (long)bz * sC;

  const int row0 = blockIdx.y * TM;
  const int col0 = blockIdx.x * TN;
  const int t    = threadIdx.x;
  const int lane = t & 31, l16 = lane & 15, hlf = lane >> 4;
  const int wave = t >> 5, wr = wave >> 1, wc = wave & 1;   // 4x2 wave grid

  // staging assignments (contiguous 16B chunks)
  const int ar = t >> 2, akb = (t & 3) * 8;     // A: 64 rows x 32k, 8 elems/thread
  const int br = t >> 1, bkb = (t & 1) * 16;    // B: 128 rows x 32k, 16 elems/thread
  const __bf16* aG = A  + (long)(row0 + ar) * lda + akb;
  const __bf16* bG = Bt + (long)(col0 + br) * ldb + bkb;

  v8f acc[4] = {{}, {}, {}, {}};

#if USE_ASYNC_LDS
  // ---- double-buffered LDS, async global->LDS copies (ASYNCcnt) ----
  __shared__ __bf16 lds_a[2][TM * LDP];
  __shared__ __bf16 lds_b[2][TN * LDP];
  __bf16* la0 = lds_a[0] + ar * LDP + akb;
  __bf16* lb0 = lds_b[0] + br * LDP + bkb;
  __bf16* la1 = lds_a[1] + ar * LDP + akb;
  __bf16* lb1 = lds_b[1] + br * LDP + bkb;

  async_b128(aG,     la0);
  async_b128(bG,     lb0);
  async_b128(bG + 8, lb0 + 8);

  int cur = 0;
  for (int k0 = 0; k0 < K; k0 += TK, cur ^= 1) {
    WAIT_ASYNC0();        // current tile's copies landed in LDS
    __syncthreads();      // ... in every wave; prior tile's readers are done
    if (k0 + TK < K) {    // stream next tile into the other buffer
      __bf16* na = cur ? la0 : la1;
      __bf16* nb = cur ? lb0 : lb1;
      async_b128(aG + k0 + TK,     na);
      async_b128(bG + k0 + TK,     nb);
      async_b128(bG + k0 + TK + 8, nb + 8);
    }
    mma_step(lds_a[cur], lds_b[cur], wr, wc, l16, hlf, acc);
  }
#else
  // ---- register-pipelined staging fallback ----
  __shared__ __bf16 lds_a[TM * LDP];
  __shared__ __bf16 lds_b[TN * LDP];

  uint4 aR  = *(const uint4*)(aG);
  uint4 bR0 = *(const uint4*)(bG);
  uint4 bR1 = *(const uint4*)(bG + 8);

  for (int k0 = 0; k0 < K; k0 += TK) {
    *(uint4*)(lds_a + ar * LDP + akb)     = aR;
    *(uint4*)(lds_b + br * LDP + bkb)     = bR0;
    *(uint4*)(lds_b + br * LDP + bkb + 8) = bR1;
    __syncthreads();
    if (k0 + TK < K) {
      aR  = *(const uint4*)(aG + k0 + TK);
      bR0 = *(const uint4*)(bG + k0 + TK);
      bR1 = *(const uint4*)(bG + k0 + TK + 8);
    }
    mma_step(lds_a, lds_b, wr, wc, l16, hlf, acc);
    __syncthreads();
  }
#endif

  // -------- epilogue: D layout M = v + 8*half, N = lane%16 --------
  #pragma unroll
  for (int ns = 0; ns < 4; ++ns) {
    const int col = col0 + wc * 64 + ns * 16 + l16;
    float bval = 0.0f;
    if constexpr (HAS_BIAS) bval = bias[col];
    #pragma unroll
    for (int v = 0; v < 8; ++v) {
      const int row = row0 + wr * 16 + hlf * 8 + v;
      const long idx = (long)row * N + col;
      float val = acc[ns][v];
      if constexpr (SCORES) {
        // faithful to masked_fill_(mask, 1e-9)
        Cf[cOff + idx] = (col > row) ? 1e-9f : val * scale;
      } else {
        val += bval;
        if constexpr (HAS_RES)  val += res[idx];
        if constexpr (OUT_F32)  Cf[cOff + idx] = val;
        if constexpr (OUT_BF)   Cbf[cOff + idx] = (__bf16)val;
        if constexpr (OUT_BFT)  Cbf[(long)col * ldct + row] = (__bf16)val;  // V^T
      }
    }
  }
}

// ---------------- row softmax over S=2048; fp32 out + bf16 copy ------------
__global__ __launch_bounds__(256)
void softmax_rows(float* __restrict__ attn_f, __bf16* __restrict__ attn_bf, int S)
{
  __shared__ float red[256];
  const long row = blockIdx.x;
  float* p = attn_f + row * S;
  const int t = threadIdx.x;

  float vals[8];
  float m = -1e30f;
  #pragma unroll
  for (int j = 0; j < 8; ++j) { vals[j] = p[t + j * 256]; m = fmaxf(m, vals[j]); }
  red[t] = m; __syncthreads();
  for (int s = 128; s > 0; s >>= 1) { if (t < s) red[t] = fmaxf(red[t], red[t + s]); __syncthreads(); }
  m = red[0]; __syncthreads();

  float sum = 0.0f;
  #pragma unroll
  for (int j = 0; j < 8; ++j) { vals[j] = __expf(vals[j] - m); sum += vals[j]; }
  red[t] = sum; __syncthreads();
  for (int s = 128; s > 0; s >>= 1) { if (t < s) red[t] += red[t + s]; __syncthreads(); }
  const float inv = 1.0f / red[0];

  #pragma unroll
  for (int j = 0; j < 8; ++j) {
    const float a = vals[j] * inv;
    p[t + j * 256] = a;
    attn_bf[row * S + t + j * 256] = (__bf16)a;
  }
}

// ---------------- in-place LayerNorm over rows of D=1024 -------------------
__global__ __launch_bounds__(256)
void layernorm_rows(float* __restrict__ y, const float* __restrict__ gamma,
                    const float* __restrict__ beta, int Dd)
{
  __shared__ float r1[256];
  __shared__ float r2[256];
  const long row = blockIdx.x;
  float* p = y + row * Dd;
  const int t = threadIdx.x;

  float v[4]; float s = 0.0f, s2 = 0.0f;
  #pragma unroll
  for (int j = 0; j < 4; ++j) { v[j] = p[t + j * 256]; s += v[j]; s2 += v[j] * v[j]; }
  r1[t] = s; r2[t] = s2; __syncthreads();
  for (int k = 128; k > 0; k >>= 1) {
    if (t < k) { r1[t] += r1[t + k]; r2[t] += r2[t + k]; }
    __syncthreads();
  }
  const float mu  = r1[0] / (float)Dd;
  const float var = r2[0] / (float)Dd - mu * mu;
  const float rs  = rsqrtf(var + 1e-5f);
  #pragma unroll
  for (int j = 0; j < 4; ++j) {
    const int c = t + j * 256;
    p[c] = (v[j] - mu) * rs * gamma[c] + beta[c];
  }
}

// ---------------------------------------------------------------------------
extern "C" void kernel_launch(void* const* d_in, const int* in_sizes, int n_in,
                              void* d_out, int out_size, void* d_ws, size_t ws_size,
                              hipStream_t stream) {
  (void)in_sizes; (void)n_in; (void)out_size; (void)ws_size;

  constexpr int Bv = 4, Sv = 2048, Dv = 1024, HDv = 1024;
  constexpr int Mall = Bv * Sv;               // 8192
  constexpr long nX = (long)Mall * Dv;        // 8,388,608
  constexpr long nW = (long)Dv * HDv;         // 1,048,576
  constexpr long nAttn = (long)Bv * Sv * Sv;  // 16,777,216

  const float* x     = (const float*)d_in[0];
  const float* wq    = (const float*)d_in[2];
  const float* bq    = (const float*)d_in[3];
  const float* wk    = (const float*)d_in[4];
  const float* bk    = (const float*)d_in[5];
  const float* wv    = (const float*)d_in[6];
  const float* bv    = (const float*)d_in[7];
  const float* wo    = (const float*)d_in[8];
  const float* bo    = (const float*)d_in[9];
  const float* gamma = (const float*)d_in[10];
  const float* beta  = (const float*)d_in[11];

  float* out_y    = (float*)d_out;        // [B,S,D]
  float* out_attn = out_y + nX;           // [B,S,S]

  char* ws = (char*)d_ws;
  __bf16* x_bf   = (__bf16*)ws; ws += nX * 2;     // x, bf16 [8192,1024]
  __bf16* wqT_bf = (__bf16*)ws; ws += nW * 2;     // wq^T [HD,D]
  __bf16* wkT_bf = (__bf16*)ws; ws += nW * 2;
  __bf16* wvT_bf = (__bf16*)ws; ws += nW * 2;
  __bf16* woT_bf = (__bf16*)ws; ws += nW * 2;     // wo^T [D,HD]
  __bf16* q_bf   = (__bf16*)ws; ws += nX * 2;     // Q [8192,1024]
  __bf16* k_bf   = (__bf16*)ws; ws += nX * 2;     // K [8192,1024]
  __bf16* vT_bf  = (__bf16*)ws; ws += nX * 2;     // V^T [1024, 8192]
  __bf16* a_bf   = (__bf16*)ws; ws += nAttn * 2;  // attn bf16 [B,S,S]
  __bf16* c_bf   = (__bf16*)ws; ws += nX * 2;     // context+resid [8192,1024]

  const dim3 blk(256);

  // 1) downconvert: x plain, weights transposed (once; hot loops stay NT)
  cvt_f32_bf16<<<(nX + 255) / 256, blk, 0, stream>>>(x, x_bf, nX);
  {
    const dim3 g(Dv / 32, HDv / 32);
    cvt_transpose_bf16<<<g, blk, 0, stream>>>(wq, wqT_bf, Dv, HDv);
    cvt_transpose_bf16<<<g, blk, 0, stream>>>(wk, wkT_bf, Dv, HDv);
    cvt_transpose_bf16<<<g, blk, 0, stream>>>(wv, wvT_bf, Dv, HDv);
    cvt_transpose_bf16<<<g, blk, 0, stream>>>(wo, woT_bf, HDv, Dv);
  }

  // 2) Q/K/V projections: [8192,1024] x [1024,1024]^T (+bias)
  {
    const dim3 grid(HDv / TN, Mall / TM, 1);
    gemm_wmma<true, false, true, false, false, false><<<grid, blk, 0, stream>>>(
        x_bf, wqT_bf, bq, nullptr, q_bf, nullptr,
        Mall, HDv, Dv, Dv, Dv, 0, 0, 0, 0, 0, 1.0f);
    gemm_wmma<true, false, true, false, false, false><<<grid, blk, 0, stream>>>(
        x_bf, wkT_bf, bk, nullptr, k_bf, nullptr,
        Mall, HDv, Dv, Dv, Dv, 0, 0, 0, 0, 0, 1.0f);
    // V stored transposed (ldct = 8192) so the context GEMM is NT too
    gemm_wmma<true, false, false, true, false, false><<<grid, blk, 0, stream>>>(
        x_bf, wvT_bf, bv, nullptr, vT_bf, nullptr,
        Mall, HDv, Dv, Dv, Dv, Mall, 0, 0, 0, 0, 1.0f);
  }

  // 3) scores = Q K^T / sqrt(64), causal masked-fill(1e-9), per batch
  {
    const dim3 grid(Sv / TN, Sv / TM, Bv);
    gemm_wmma<false, false, false, false, false, true><<<grid, blk, 0, stream>>>(
        q_bf, k_bf, nullptr, nullptr, nullptr, out_attn,
        Sv, Sv, HDv, HDv, HDv, 0,
        (long)Sv * HDv, (long)Sv * HDv, 0, (long)Sv * Sv, 0.125f);
  }

  // 4) softmax rows -> fp32 attn output + bf16 copy for context GEMM
  softmax_rows<<<Bv * Sv, blk, 0, stream>>>(out_attn, a_bf, Sv);

  // 5) context = attn @ V + x (residual): attn[S,S] x V^T[HD, B*S] per batch
  {
    const dim3 grid(HDv / TN, Sv / TM, Bv);
    gemm_wmma<false, true, true, false, false, false><<<grid, blk, 0, stream>>>(
        a_bf, vT_bf, nullptr, x, c_bf, nullptr,
        Sv, HDv, Sv, Sv, Mall, 0,
        (long)Sv * Sv, (long)Sv, (long)Sv * Dv, (long)Sv * HDv, 1.0f);
  }

  // 6) output projection -> fp32 into d_out
  {
    const dim3 grid(Dv / TN, Mall / TM, 1);
    gemm_wmma<true, false, false, false, true, false><<<grid, blk, 0, stream>>>(
        c_bf, woT_bf, bo, nullptr, nullptr, out_y,
        Mall, Dv, HDv, HDv, HDv, 0, 0, 0, 0, 0, 1.0f);
  }

  // 7) LayerNorm in-place on d_out rows
  layernorm_rows<<<Bv * Sv, blk, 0, stream>>>(out_y, gamma, beta, Dv);
}
